// MaSST_90555090468989
// MI455X (gfx1250) — compile-verified, matrix operand order checked by reading, not run on hardware
//
#include <hip/hip_runtime.h>

// MARNN forward for MI455X (gfx1250, wave32).
//  - All GEMMs: v_wmma_f32_16x16x32_f16, f32 accumulate.
//  - B (weights) pre-packed once per launch into the WMMA per-lane f16 layout
//    -> hot loop reads B with coalesced global_load_b128, prefetched 1 ahead.
//  - A double-buffered in LDS via CDNA5 global_load_async_to_lds_b128
//    (ASYNCcnt, in-order completion => s_wait_asynccnt 4 overlaps next tile),
//    read back as 4x ds_load_b128 in WMMA register order, cvt_pk to f16.
//  - Steady-state k-loop is branchless with strength-reduced addressing and
//    2x unroll so A/B registers ping-pong instead of being copied.
//  - Shape-compatible GEMM pairs (i/h MLP, wi/wh GRU, gi/gh aux) batched into
//    one launch via blockIdx.y.

typedef __attribute__((ext_vector_type(16))) _Float16 v16h;
typedef __attribute__((ext_vector_type(8)))  float    v8f;

#define TT    128
#define BB    32
#define FF    500
#define HH    512
#define MM    64
#define HEADN 64
#define AA    48
#define AUXH  130      // 2*(HEAD+1)
#define AUXP  160      // AUXH padded to multiple of 32
#define NEGV  (-9999999.0f)

__device__ __forceinline__ float fsigmoid(float x) { return 1.0f / (1.0f + __expf(-x)); }

// ---------------------------------------------------------------------------
// Weight packer: f32 W[K,N] -> f16 tiles Wp[ntile][kblock][512] where the 512
// halves are in the exact per-lane order of the WMMA B operand (32x16 tile):
//   element (kk,n) -> lane = (n&15) + 16*((kk>>3)&1), slot j = (kk&7)+8*(kk>>4)
// Zero-fills K/N padding, so the GEMM needs no bounds checks on B.
// ---------------------------------------------------------------------------
__global__ void k_pack_w(const float* __restrict__ W, int K, int N,
                         _Float16* __restrict__ Wp)
{
  const int KB     = (K + 31) >> 5;
  const int tilesN = (N + 15) >> 4;
  const int total  = tilesN * KB * 512;
  const int i = blockIdx.x * blockDim.x + threadIdx.x;
  if (i >= total) return;
  const int pos  = i & 511;
  const int kb   = (i >> 9) % KB;
  const int nt   = (i >> 9) / KB;
  const int lane = pos >> 4, j = pos & 15;
  const int n  = (nt << 4) + (lane & 15);
  const int kk = (j & 7) | ((lane >> 4) << 3) | ((j >> 3) << 4);
  const int k  = (kb << 5) + kk;
  Wp[i] = (k < K && n < N) ? (_Float16)W[k * N + n] : (_Float16)0.f;
}

// ---------------------------------------------------------------------------
// Batched GEMM: for set s = blockIdx.y,
//   C[s][M,N] = act( [A0[s]|A1[s]][M, Ka+Kb] @ Wp[s] + bias[s] )
// Requirements: M % 16 == 0, Ka % 32 == 0, Kb % 32 == 0 (pad A with zeros).
// One wave per 16x16 C tile; A tile double-buffered via async-to-LDS.
// ---------------------------------------------------------------------------
struct GemmPairArgs {
  const float*    A0[2];
  const float*    A1[2];
  const _Float16* Wp[2];
  const float*    bias[2];
  float*          C[2];
  int             Ka[2];
  int             Kb[2];
  int N, ldc, M, act;
};

__global__ __launch_bounds__(256) void wmma_gemm_f16(GemmPairArgs P)
{
  __shared__ __align__(16) float smemf[8 * 1024];   // 8 waves * 2 x (16x32 f32)
  const int s    = blockIdx.y;
  const float* A0 = P.A0[s];
  const float* A1 = P.A1[s] ? P.A1[s] : P.A0[s];
  const _Float16* Wp = P.Wp[s];
  const float* bias  = P.bias[s];
  float* C = P.C[s];
  const int Ka = P.Ka[s], Kb = P.Kb[s];
  const int N = P.N, ldc = P.ldc, M = P.M, act = P.act;

  const int lane = threadIdx.x & 31;
  const int wv   = threadIdx.x >> 5;
  const int wave = blockIdx.x * 8 + wv;

  const int tilesN = (N + 15) >> 4;
  const bool active = wave < (M >> 4) * tilesN;
  const int tile  = active ? wave : 0;
  const int row0  = (tile / tilesN) << 4;
  const int ntile = tile % tilesN;
  const int KaB = Ka >> 5;
  const int KB  = (Ka + Kb) >> 5;

  float* buf0 = smemf + wv * 1024;
  float* buf1 = buf0 + 512;
  const _Float16* wp = Wp + (size_t)ntile * KB * 512;

  const int m_out  = lane & 15;          // row this lane owns in WMMA A layout
  const int kh_out = (lane >> 4) << 3;   // k-half offset (0 or 8)

  // Strength-reduced async staging: lane covers quads g = i*32+lane of the
  // 16x32 tile (row r0+4i, cols q*4..). Source = per-region base + kb*128B
  // + i*(16*lda)B; LDS dest = base + i*512B. No multiplies in the loop.
  const int r0 = lane >> 3;              // 0..3
  const int q4 = (lane & 7) << 2;        // column offset in floats
  const char* a0base = (const char*)(A0 + (size_t)(row0 + r0) * Ka + q4);
  const char* a1base = (const char*)(A1 + (size_t)(row0 + r0) * Kb + q4);
  const size_t rs0 = (size_t)Ka * 16;    // 4 rows in bytes (region 0)
  const size_t rs1 = (size_t)Kb * 16;    // 4 rows in bytes (region 1)
  const unsigned int dst0 = (unsigned int)(uintptr_t)(buf0 + r0 * 32 + q4);
  const unsigned int dst1 = (unsigned int)(uintptr_t)(buf1 + r0 * 32 + q4);

  auto issueA = [&](int kb, unsigned int dst) {
    const char* src; size_t rstep;
    if (kb < KaB) { src = a0base + (size_t)kb * 128;         rstep = rs0; }
    else          { src = a1base + (size_t)(kb - KaB) * 128; rstep = rs1; }
#pragma unroll
    for (int i = 0; i < 4; ++i) {
      asm volatile("global_load_async_to_lds_b128 %0, %1, off"
                   :: "v"(dst + (unsigned int)i * 512u),
                      "v"((unsigned long long)(uintptr_t)(src + i * rstep))
                   : "memory");
    }
  };

  auto loadA = [&](const float* buf) -> v16h {
    const float* abase = buf + m_out * 32 + kh_out;
    const float4 p0 = *(const float4*)(abase + 0);
    const float4 p1 = *(const float4*)(abase + 4);
    const float4 p2 = *(const float4*)(abase + 16);
    const float4 p3 = *(const float4*)(abase + 20);
    v16h a;
    a[0]  = (_Float16)p0.x; a[1]  = (_Float16)p0.y; a[2]  = (_Float16)p0.z; a[3]  = (_Float16)p0.w;
    a[4]  = (_Float16)p1.x; a[5]  = (_Float16)p1.y; a[6]  = (_Float16)p1.z; a[7]  = (_Float16)p1.w;
    a[8]  = (_Float16)p2.x; a[9]  = (_Float16)p2.y; a[10] = (_Float16)p2.z; a[11] = (_Float16)p2.w;
    a[12] = (_Float16)p3.x; a[13] = (_Float16)p3.y; a[14] = (_Float16)p3.z; a[15] = (_Float16)p3.w;
    return a;
  };

  v8f acc = {};
  issueA(0, dst0);
  v16h bcur = *(const v16h*)&wp[lane * 16];   // prefetched B block 0

  int kb = 0;
  // Steady state: branchless, double-buffered; 2x unroll ping-pongs registers.
#pragma unroll 2
  for (; kb + 1 < KB; ++kb) {
    const bool odd = (kb & 1) != 0;
    // WAR guard: our previous ds_loads of the other buffer must be done before
    // the async engine overwrites it (ASYNC vs DS is unordered).
    asm volatile("s_wait_dscnt 0x0" ::: "memory");
    issueA(kb + 1, odd ? dst0 : dst1);
    // ASYNCcnt completes in order: <=4 outstanding proves current tile landed.
    asm volatile("s_wait_asynccnt 0x4" ::: "memory");
    const v16h a = loadA(odd ? buf1 : buf0);
    const v16h buse = bcur;
    bcur = *(const v16h*)&wp[(size_t)(kb + 1) * 512 + lane * 16];
    acc = __builtin_amdgcn_wmma_f32_16x16x32_f16(false, a, false, buse, (short)0,
                                                 acc, false, false);
  }
  // Last k-block.
  asm volatile("s_wait_asynccnt 0x0" ::: "memory");
  {
    const v16h a = loadA((kb & 1) ? buf1 : buf0);
    acc = __builtin_amdgcn_wmma_f32_16x16x32_f16(false, a, false, bcur, (short)0,
                                                 acc, false, false);
  }

  if (active) {
    const int n = (ntile << 4) + (lane & 15);
    if (n < N) {
      const float bv = bias[n];
#pragma unroll
      for (int r = 0; r < 8; ++r) {
        const int m = row0 + r + ((lane >> 4) << 3);
        float v = acc[r] + bv;
        if (act == 1) v = fmaxf(v, 0.f);
        C[(size_t)m * ldc + n] = v;
      }
    }
  }
}

// ---------------------------------------------------------------------------
// Elementwise / control kernels
// ---------------------------------------------------------------------------
__global__ void k_xpad(const float* __restrict__ x, float* __restrict__ x_pad)
{
  const int i = blockIdx.x * blockDim.x + threadIdx.x;   // over 4096*512
  if (i >= TT * BB * HH) return;
  const int r = i >> 9, c = i & 511;
  x_pad[i] = (c < FF) ? x[r * FF + c] : 0.f;
}

__global__ void k_init_state(float* h, float* aux_h, float* imem, float* hmem,
                             float* ilu, float* hlu)
{
  const int i = blockIdx.x * blockDim.x + threadIdx.x;
  if (i < BB * MM * HH) { imem[i] = 0.f; hmem[i] = 0.f; }
  if (i < BB * HH)   h[i] = 0.f;
  if (i < BB * AUXP) aux_h[i] = 0.f;     // includes the K-padding columns
  if (i < BB * MM)  { ilu[i] = NEGV; hlu[i] = NEGV; }
}

// aux GRU gates from gi=[x_t|h]@aux_wih+bih, gh=aux_h@aux_whh+bhh (both [32,390]).
__global__ void k_aux_gates(const float* __restrict__ gi, const float* __restrict__ gh,
                            float* __restrict__ aux_h, float* __restrict__ read_head,
                            const int* __restrict__ length, int t)
{
  const int i = blockIdx.x * blockDim.x + threadIdx.x;
  if (i >= BB * AUXH) return;
  const int b = i / AUXH, j = i % AUXH;
  const float* gib = gi + b * 390;
  const float* ghb = gh + b * 390;
  const float r = fsigmoid(gib[j] + ghb[j]);
  const float z = fsigmoid(gib[AUXH + j] + ghb[AUXH + j]);
  const float n = tanhf(gib[2 * AUXH + j] + r * ghb[2 * AUXH + j]);
  const float hp = aux_h[b * AUXP + j];
  const float rh = (1.f - z) * n + z * hp;
  read_head[i] = rh;                                // unmasked value feeds scores
  aux_h[b * AUXP + j] = (t < length[b]) ? rh : hp;  // masked carry update
}

// MLP layer 2 (sigmoid) fused with the score reduction; w2 staged in LDS.
// blockIdx.y selects the i-memory (0) or h-memory (1) parameter set.
struct ScorePairArgs {
  const float* t1[2];
  const float* w2[2];
  const float* b2[2];
  const float* lu[2];
  float*       score[2];
  int          rh_off[2];
};

__global__ __launch_bounds__(256) void k_mlp2_score(ScorePairArgs S,
                                                    const float* __restrict__ read_head)
{
  const int sidx = blockIdx.y;
  const float* t1 = S.t1[sidx];
  const float* w2 = S.w2[sidx];
  const float* b2 = S.b2[sidx];
  const float* lu = S.lu[sidx];
  float* score    = S.score[sidx];
  const int rh_off = S.rh_off[sidx];

  __shared__ float sw2[32 * HEADN];
  __shared__ float sb2[HEADN];
  for (int p = threadIdx.x; p < 32 * HEADN; p += 256) sw2[p] = w2[p];
  if (threadIdx.x < HEADN) sb2[threadIdx.x] = b2[threadIdx.x];
  __syncthreads();

  const int i = blockIdx.x * 256 + threadIdx.x;   // over B*M rows
  if (i >= BB * MM) return;
  const int b = i >> 6;
  const float* row = t1 + i * 32;
  const float* rh  = read_head + b * AUXH + rh_off;
  float t1r[32];
#pragma unroll
  for (int k = 0; k < 32; ++k) t1r[k] = row[k];
  float s = 0.f;
  for (int j = 0; j < HEADN; ++j) {
    float acc = sb2[j];
#pragma unroll
    for (int k = 0; k < 32; ++k) acc += t1r[k] * sw2[k * HEADN + j];
    s += fsigmoid(acc) * rh[j];
  }
  s += fsigmoid(lu[i]) * rh[HEADN];
  score[i] = s;
}

// Gumbel-ST forward value = hard one-hot of argmax(score + noise) (tau>0).
// idx layout: [rm_i(32) | rm_h(32) | w_i(32) | w_h(32)]
__global__ void k_argmax(const float* __restrict__ sc_i, const float* __restrict__ sc_h,
                         const float* __restrict__ gum_i, const float* __restrict__ gum_h,
                         int* __restrict__ idx, int t)
{
  const int b = threadIdx.x;
  if (b >= BB) return;
  const float* gi = gum_i + (t * BB + b) * MM;
  const float* gh = gum_h + (t * BB + b) * MM;
  float bi = -1e30f, bh = -1e30f;
  int ri = 0, rhh = 0;
  for (int m = 0; m < MM; ++m) {
    const float vi = sc_i[b * MM + m] + gi[m];
    const float vh = sc_h[b * MM + m] + gh[m];
    if (vi > bi) { bi = vi; ri = m; }
    if (vh > bh) { bh = vh; rhh = m; }
  }
  idx[b]          = ri;
  idx[BB + b]     = rhh;
  idx[2 * BB + b] = (t < MM) ? t : ri;
  idx[3 * BB + b] = (t < MM) ? t : rhh;
}

// One-hot read (entry gather), one-hot write (slot overwrite), lu update.
__global__ void k_entry_memwrite(const int* __restrict__ idx,
                                 float* __restrict__ imem, float* __restrict__ hmem,
                                 const float* __restrict__ xp_t, const float* __restrict__ h,
                                 float* __restrict__ i_entry, float* __restrict__ h_entry,
                                 float* __restrict__ ilu, float* __restrict__ hlu)
{
  const int i = blockIdx.x * blockDim.x + threadIdx.x;
  if (i < BB * MM) {   // (lu-1)*(1-onehot): selected slot -> 0, others -> v-1
    const int b = i >> 6, m = i & 63;
    const float vi = ilu[i];
    ilu[i] = (m == idx[b]) ? 0.f : (vi - 1.f);
    const float vh = hlu[i];
    hlu[i] = (m == idx[BB + b]) ? 0.f : (vh - 1.f);
  }
  if (i < BB * HH) {
    const int b = i >> 9, c = i & 511;
    const int rmi = idx[b], rmh = idx[BB + b];
    const int wi = idx[2 * BB + b], wh = idx[3 * BB + b];
    const float ie = imem[(b * MM + rmi) * HH + c];
    const float he = hmem[(b * MM + rmh) * HH + c];
    i_entry[b * HH + c] = ie;
    h_entry[b * HH + c] = he;
    imem[(b * MM + wi) * HH + c] = xp_t[b * HH + c];
    hmem[(b * MM + wh) * HH + c] = h[b * HH + c];   // old h (updated later)
  }
}

__global__ void k_gru_update(const float* __restrict__ wi, const float* __restrict__ wh,
                             float* __restrict__ h, float* __restrict__ ys,
                             const int* __restrict__ length, int t)
{
  const int i = blockIdx.x * blockDim.x + threadIdx.x;
  if (i >= BB * HH) return;
  const int b = i >> 9, j = i & 511;
  const float* wib = wi + b * 1536;
  const float* whb = wh + b * 1536;
  const float r = fsigmoid(wib[j] + whb[j]);
  const float z = fsigmoid(wib[HH + j] + whb[HH + j]);
  const float n = tanhf(wib[2 * HH + j] + r * whb[2 * HH + j]);
  const float hp = h[i];
  const float h1 = (1.f - z) * n + z * hp;
  const float hn = (t < length[b]) ? h1 : hp;
  h[i] = hn;
  ys[(t * BB + b) * HH + j] = hn;
}

// ---------------------------------------------------------------------------
extern "C" void kernel_launch(void* const* d_in, const int* in_sizes, int n_in,
                              void* d_out, int out_size, void* d_ws, size_t ws_size,
                              hipStream_t stream)
{
  (void)in_sizes; (void)n_in; (void)out_size; (void)ws_size;
  const float* x       = (const float*)d_in[0];
  const int*   length  = (const int*)d_in[1];
  const float* fc1_w   = (const float*)d_in[2];
  const float* fc1_b   = (const float*)d_in[3];
  const float* w_ih    = (const float*)d_in[4];
  const float* w_hh    = (const float*)d_in[5];
  const float* b_ih    = (const float*)d_in[6];
  const float* b_hh    = (const float*)d_in[7];
  const float* aux_wih = (const float*)d_in[8];
  const float* aux_whh = (const float*)d_in[9];
  const float* aux_bih = (const float*)d_in[10];
  const float* aux_bhh = (const float*)d_in[11];
  const float* ifc_w1  = (const float*)d_in[12];
  const float* ifc_b1  = (const float*)d_in[13];
  const float* ifc_w2  = (const float*)d_in[14];
  const float* ifc_b2  = (const float*)d_in[15];
  const float* hfc_w1  = (const float*)d_in[16];
  const float* hfc_b1  = (const float*)d_in[17];
  const float* hfc_w2  = (const float*)d_in[18];
  const float* hfc_b2  = (const float*)d_in[19];
  const float* fc_w    = (const float*)d_in[20];
  const float* fc_b    = (const float*)d_in[21];
  const float* gum_i   = (const float*)d_in[22];
  const float* gum_h   = (const float*)d_in[23];
  float* out = (float*)d_out;

  float* ws = (float*)d_ws;
  size_t off = 0;
  auto alloc  = [&](size_t n) { float* p = ws + off; off += n; return p; };
  auto allocH = [&](size_t nh) { float* p = ws + off; off += (nh + 1) / 2; return (_Float16*)p; };

  float* x_pad     = alloc((size_t)TT * BB * HH);
  float* xp        = alloc((size_t)TT * BB * HH);
  float* ys        = alloc((size_t)TT * BB * HH);
  float* h         = alloc(BB * HH);
  float* aux_h     = alloc(BB * AUXP);
  float* imem      = alloc((size_t)BB * MM * HH);
  float* hmem      = alloc((size_t)BB * MM * HH);
  float* ilu       = alloc(BB * MM);
  float* hlu       = alloc(BB * MM);
  float* gi        = alloc(BB * 390);
  float* gh        = alloc(BB * 390);
  float* read_head = alloc(BB * AUXH);
  float* t1_i      = alloc(BB * MM * 32);
  float* t1_h      = alloc(BB * MM * 32);
  float* sc_i      = alloc(BB * MM);
  float* sc_h      = alloc(BB * MM);
  float* i_entry   = alloc(BB * HH);
  float* h_entry   = alloc(BB * HH);
  float* wi        = alloc(BB * 1536);
  float* wh        = alloc(BB * 1536);
  int*   idx       = (int*)alloc(4 * BB);

  // Packed weights (f16), tilesN * ceil(K/32) * 512 halves each.
  _Float16* p_fc1     = allocH((size_t)32 * 16 * 512);   // [500->512, 512]
  _Float16* p_aux_wih = allocH((size_t)25 * 32 * 512);   // [1024, 390]
  _Float16* p_aux_whh = allocH((size_t)25 *  5 * 512);   // [130->160, 390]
  _Float16* p_ifc_w1  = allocH((size_t) 2 * 16 * 512);   // [512, 32]
  _Float16* p_hfc_w1  = allocH((size_t) 2 * 16 * 512);
  _Float16* p_w_ih    = allocH((size_t)96 * 32 * 512);   // [1024, 1536]
  _Float16* p_w_hh    = allocH((size_t)96 * 32 * 512);
  _Float16* p_fc_w    = allocH((size_t) 3 * 16 * 512);   // [512, 48]

  auto pack = [&](const float* W, int K, int N, _Float16* Wp) {
    const int total = ((N + 15) / 16) * ((K + 31) / 32) * 512;
    k_pack_w<<<(total + 255) / 256, 256, 0, stream>>>(W, K, N, Wp);
  };
  auto gemm2 = [&](int nsets,
                   const float* A0a, int Kaa, const float* A1a, int Kba,
                   const _Float16* Wpa, const float* ba, float* Ca,
                   const float* A0b, int Kab, const float* A1b, int Kbb,
                   const _Float16* Wpb, const float* bb, float* Cb,
                   int N, int ldc, int M, int act) {
    GemmPairArgs P;
    P.A0[0] = A0a; P.A1[0] = A1a; P.Wp[0] = Wpa; P.bias[0] = ba; P.C[0] = Ca;
    P.Ka[0] = Kaa; P.Kb[0] = Kba;
    P.A0[1] = A0b; P.A1[1] = A1b; P.Wp[1] = Wpb; P.bias[1] = bb; P.C[1] = Cb;
    P.Ka[1] = Kab; P.Kb[1] = Kbb;
    P.N = N; P.ldc = ldc; P.M = M; P.act = act;
    const int waves  = (M / 16) * ((N + 15) / 16);
    const int blocks = (waves + 7) / 8;   // 8 waves (256 threads) per block
    wmma_gemm_f16<<<dim3(blocks, nsets, 1), 256, 0, stream>>>(P);
  };

  k_init_state<<<(BB * MM * HH + 255) / 256, 256, 0, stream>>>(h, aux_h, imem, hmem, ilu, hlu);
  k_xpad<<<(TT * BB * HH + 255) / 256, 256, 0, stream>>>(x, x_pad);

  pack(fc1_w,   FF,   HH,     p_fc1);
  pack(aux_wih, 2*HH, 3*AUXH, p_aux_wih);
  pack(aux_whh, AUXH, 3*AUXH, p_aux_whh);
  pack(ifc_w1,  HH,   32,     p_ifc_w1);
  pack(hfc_w1,  HH,   32,     p_hfc_w1);
  pack(w_ih,    2*HH, 3*HH,   p_w_ih);
  pack(w_hh,    2*HH, 3*HH,   p_w_hh);
  pack(fc_w,    HH,   AA,     p_fc_w);

  // xp = x @ fc1_w + fc1_b     [4096, 512pad] @ [512pad, 512]
  gemm2(1, x_pad, HH, nullptr, 0, p_fc1, fc1_b, xp,
           x_pad, HH, nullptr, 0, p_fc1, fc1_b, xp,
        HH, HH, TT * BB, 0);

  for (int t = 0; t < TT; ++t) {
    const float* xp_t = xp + (size_t)t * BB * HH;

    // aux GRU: gi = [x_t|h] @ aux_wih + bih ; gh = aux_h(pad) @ aux_whh + bhh
    gemm2(2, xp_t, HH, h, HH,         p_aux_wih, aux_bih, gi,
             aux_h, AUXP, nullptr, 0, p_aux_whh, aux_bhh, gh,
          3 * AUXH, 390, BB, 0);
    k_aux_gates<<<(BB * AUXH + 255) / 256, 256, 0, stream>>>(gi, gh, aux_h, read_head, length, t);

    // memory-MLP layer 1 (ReLU): [2048, 512] @ [512, 32], i & h batched
    gemm2(2, imem, HH, nullptr, 0, p_ifc_w1, ifc_b1, t1_i,
             hmem, HH, nullptr, 0, p_hfc_w1, hfc_b1, t1_h,
          32, 32, BB * MM, 1);

    {
      ScorePairArgs S;
      S.t1[0] = t1_i; S.w2[0] = ifc_w2; S.b2[0] = ifc_b2; S.lu[0] = ilu;
      S.score[0] = sc_i; S.rh_off[0] = 0;
      S.t1[1] = t1_h; S.w2[1] = hfc_w2; S.b2[1] = hfc_b2; S.lu[1] = hlu;
      S.score[1] = sc_h; S.rh_off[1] = HEADN + 1;
      k_mlp2_score<<<dim3((BB * MM + 255) / 256, 2, 1), 256, 0, stream>>>(S, read_head);
    }

    k_argmax<<<1, 32, 0, stream>>>(sc_i, sc_h, gum_i, gum_h, idx, t);
    k_entry_memwrite<<<(BB * HH + 255) / 256, 256, 0, stream>>>(idx, imem, hmem, xp_t, h,
                                                                i_entry, h_entry, ilu, hlu);

    // main GRU: wi = [x_t|i_entry] @ w_ih + b_ih ; wh = [h|h_entry] @ w_hh + b_hh
    gemm2(2, xp_t, HH, i_entry, HH, p_w_ih, b_ih, wi,
             h,    HH, h_entry, HH, p_w_hh, b_hh, wh,
          3 * HH, 3 * HH, BB, 0);
    k_gru_update<<<(BB * HH + 255) / 256, 256, 0, stream>>>(wi, wh, h, ys, length, t);
  }

  // out = ys @ fc_w + fc_b     [4096, 512] @ [512, 48]
  gemm2(1, ys, HH, nullptr, 0, p_fc_w, fc_b, out,
           ys, HH, nullptr, 0, p_fc_w, fc_b, out,
        AA, AA, TT * BB, 0);
}
